// TransformerBlock_67319317397820
// MI455X (gfx1250) — compile-verified, matrix operand order checked by reading, not run on hardware
//
#include <hip/hip_runtime.h>

// ---------------- problem constants ----------------
#define D_MODEL 2048
#define N_HEADS 16
#define D_FF    8192
#define SEQ     2048
#define BATCH   2
#define DK      128          // D_MODEL / N_HEADS
#define NTOK    (BATCH*SEQ)  // 4096 rows of activations
#define D3      (3*D_MODEL)  // 6144
#define EPSV    1e-5f

// ---------------- WMMA types ----------------
typedef __attribute__((ext_vector_type(16))) __bf16 bf16x16;
typedef __attribute__((ext_vector_type(8)))  float  v8f;
typedef __attribute__((ext_vector_type(4)))  unsigned int u32x4;
typedef __attribute__((ext_vector_type(8)))  int  i32x8;
typedef __attribute__((ext_vector_type(4)))  int  i32x4;

union FragBF {
    bf16x16      bf;
    unsigned int u[8];
};

__device__ __forceinline__ v8f wmma_bf16(const FragBF& a, const FragBF& b, v8f c) {
    return __builtin_amdgcn_wmma_f32_16x16x32_bf16(
        /*neg_a=*/false, a.bf, /*neg_b=*/false, b.bf,
        /*c_mod=*/(short)0, c, /*reuse_a=*/false, /*reuse_b=*/false);
}

// K mapping inside a 16-bit 16x32 A (or mirrored B) fragment:
// VGPR v in 0..3 -> K = 2v + 8*half ; v in 4..7 -> K = 16 + 2(v-4) + 8*half
__device__ __forceinline__ int kmap(int v, int half) {
    return (v < 4) ? (2 * v + 8 * half) : (16 + 2 * (v - 4) + 8 * half);
}

__device__ __forceinline__ unsigned short f2bf(float f) {
    union { float f; unsigned u; } v; v.f = f;
    unsigned r = v.u + 0x7fffu + ((v.u >> 16) & 1u);   // round-nearest-even
    return (unsigned short)(r >> 16);
}

// ---------------- Tensor Data Mover: 2-D bf16 tile load, HW row padding -------
// D# per CDNA5 ISA ch.8: group0 {count, lds_addr, global_addr, type=2},
// group1 {data_size=2B, pad_enable, pad_interval, pad_amount, tensor dims,
//         tile dims, dim0 stride}, groups 2/3 = 0 (2-D tile).
// This toolchain exposes the 6-arg builtin (extra zero-filled i32x8 group).
__device__ __forceinline__ void tdm_load_2d(unsigned lds_addr,
                                            const unsigned short* gsrc,
                                            int tile0 /*elems per row*/,
                                            int tile1 /*rows*/,
                                            unsigned long long stride_elems,
                                            int pad_interval /*enc: dwords*/,
                                            int pad_amount   /*enc: dwords*/,
                                            unsigned tdim0) {
    unsigned long long ga = (unsigned long long)(const void*)gsrc;
    u32x4 g0;
    g0.x = 1u;                                            // count=1, user mode
    g0.y = lds_addr;                                      // LDS byte address
    g0.z = (unsigned)(ga & 0xffffffffull);                // global_addr[31:0]
    g0.w = (unsigned)((ga >> 32) & 0x1ffffffull) | (2u << 30); // [56:32] | type=2
    i32x8 g1;
    unsigned w0 = (1u << 16)                              // data_size = 2 bytes
                | (1u << 20)                              // pad_enable
                | ((unsigned)pad_interval << 22)
                | ((unsigned)pad_amount   << 25);
    unsigned tdim1 = 1u << 20;                            // generous (no OOB)
    g1[0] = (int)w0;                                      // wg_mask=0 (no cluster)
    g1[1] = (int)((tdim0 & 0xffffu) << 16);               // tensor_dim0[15:0]
    g1[2] = (int)((tdim0 >> 16) | ((tdim1 & 0xffffu) << 16)); // dim0 hi | dim1 lo
    g1[3] = (int)((tdim1 >> 16) | ((unsigned)tile0 << 16));   // dim1 hi | tile_dim0
    g1[4] = (int)tile1;                                   // tile_dim1 (tile_dim2=0)
    g1[5] = (int)(unsigned)(stride_elems & 0xffffffffull);// tensor_dim0_stride lo
    g1[6] = (int)(unsigned)((stride_elems >> 32) & 0xffffull); // stride hi
    g1[7] = 0;
    i32x4 z4 = {0, 0, 0, 0};
    i32x8 z8 = {0, 0, 0, 0, 0, 0, 0, 0};
    __builtin_amdgcn_tensor_load_to_lds(g0, g1, z4, z4, z8, 0);
}

// ---------------- fp32 -> bf16 bulk convert ----------------
__global__ __launch_bounds__(256)
void k_f32_to_bf16(const float* __restrict__ src, unsigned short* __restrict__ dst, int n) {
    for (int i = blockIdx.x * 256 + threadIdx.x; i < n; i += gridDim.x * 256)
        dst[i] = f2bf(src[i]);
}

// ---------------- RMSNorm (one row per block) ----------------
__global__ __launch_bounds__(256)
void k_rmsnorm_bf16(const float* __restrict__ X, const float* __restrict__ g,
                    unsigned short* __restrict__ H) {
    const int row = blockIdx.x;
    const int tid = threadIdx.x;
    const float* xr = X + (size_t)row * D_MODEL;
    float ss = 0.f;
    for (int c = tid; c < D_MODEL; c += 256) { float v = xr[c]; ss += v * v; }
    __shared__ float red[256];
    red[tid] = ss; __syncthreads();
    for (int s = 128; s > 0; s >>= 1) {
        if (tid < s) red[tid] += red[tid + s];
        __syncthreads();
    }
    const float inv = rsqrtf(red[0] * (1.0f / D_MODEL) + EPSV);
    unsigned short* hr = H + (size_t)row * D_MODEL;
    for (int c = tid; c < D_MODEL; c += 256)
        hr[c] = f2bf(xr[c] * inv * g[c]);
}

// ---------------- RoPE + head packing ----------------
__global__ __launch_bounds__(256)
void k_rope_pack(const float* __restrict__ qkv, const int* __restrict__ pos,
                 unsigned short* __restrict__ qh, unsigned short* __restrict__ kh,
                 unsigned short* __restrict__ vt) {
    int idx = blockIdx.x * 256 + threadIdx.x;       // B*H*S*(DK/2) threads
    const int p  = idx & 63;  idx >>= 6;
    const int s  = idx & (SEQ - 1); idx >>= 11;
    const int h  = idx & (N_HEADS - 1); idx >>= 4;
    const int b  = idx;
    const int tok = b * SEQ + s;
    const int tp  = pos[tok];
    const float ang = (float)tp * __powf(10000.0f, -(float)p * (1.0f / 64.0f));
    float cs, sn; __sincosf(ang, &sn, &cs);
    const size_t base = (size_t)tok * D3 + h * DK + 2 * p;
    float q1 = qkv[base], q2 = qkv[base + 1];
    float k1 = qkv[base + D_MODEL], k2 = qkv[base + D_MODEL + 1];
    float v1 = qkv[base + 2 * D_MODEL], v2 = qkv[base + 2 * D_MODEL + 1];
    const size_t bh = (size_t)(b * N_HEADS + h);
    const size_t qo = (bh * SEQ + s) * DK + 2 * p;
    qh[qo]     = f2bf(q1 * cs - q2 * sn);
    qh[qo + 1] = f2bf(q2 * cs + q1 * sn);
    kh[qo]     = f2bf(k1 * cs - k2 * sn);
    kh[qo + 1] = f2bf(k2 * cs + k1 * sn);
    vt[(bh * DK + 2 * p) * SEQ + s]     = f2bf(v1);
    vt[(bh * DK + 2 * p + 1) * SEQ + s] = f2bf(v2);
}

// ---------------- generic WMMA GEMM  C[M,N] = A[M,K] * B[N,K]^T (+resid) ------
// TDM loads the 128x32 bf16 tiles (with HW padding to 40-ushort rows);
// double-buffered, DMA for tile k+1 overlaps WMMA on tile k.
template<int EPI>
__global__ __launch_bounds__(256)
void k_gemm_bf16(const unsigned short* __restrict__ A,
                 const unsigned short* __restrict__ B,
                 float* __restrict__ C,
                 const float* __restrict__ resid,
                 int M, int N, int K) {
    __shared__ __align__(16) unsigned short sA[2][128][40];   // BK=32, pad->40
    __shared__ __align__(16) unsigned short sB[2][128][40];
    const int tid  = threadIdx.x;
    const int lane = tid & 31, wid = tid >> 5;
    const int half = lane >> 4, l15 = lane & 15;
    const int wm = wid & 3, wn = wid >> 2;           // 4x2 wave grid
    const int bM = blockIdx.y * 128, bN = blockIdx.x * 128;

    const unsigned ldsA0 = (unsigned)(size_t)&sA[0][0][0];
    const unsigned ldsA1 = (unsigned)(size_t)&sA[1][0][0];
    const unsigned ldsB0 = (unsigned)(size_t)&sB[0][0][0];
    const unsigned ldsB1 = (unsigned)(size_t)&sB[1][0][0];

    v8f acc[2][4];
#pragma unroll
    for (int i = 0; i < 2; ++i)
#pragma unroll
        for (int j = 0; j < 4; ++j) acc[i][j] = (v8f)(0.0f);

    const int KT = K >> 5;

    if (wid == 0) {   // wave 0 drives the Tensor Data Mover
        tdm_load_2d(ldsA0, A + (size_t)bM * K, 32, 128, (unsigned long long)K, 3, 3, (unsigned)K);
        tdm_load_2d(ldsB0, B + (size_t)bN * K, 32, 128, (unsigned long long)K, 3, 3, (unsigned)K);
        __builtin_amdgcn_s_wait_tensorcnt(0);
    }
    __syncthreads();

    for (int kt = 0; kt < KT; ++kt) {
        const int buf = kt & 1;
        if (wid == 0 && kt + 1 < KT) {               // DMA next tile during compute
            const int k0 = (kt + 1) << 5;
            tdm_load_2d(buf ? ldsA0 : ldsA1, A + (size_t)bM * K + k0, 32, 128,
                        (unsigned long long)K, 3, 3, (unsigned)K);
            tdm_load_2d(buf ? ldsB0 : ldsB1, B + (size_t)bN * K + k0, 32, 128,
                        (unsigned long long)K, 3, 3, (unsigned)K);
        }
        FragBF af[2], bfm[4];
#pragma unroll
        for (int i = 0; i < 2; ++i) {
            const int row = wm * 32 + i * 16 + l15;
#pragma unroll
            for (int v = 0; v < 8; ++v)
                af[i].u[v] = *(const unsigned int*)&sA[buf][row][kmap(v, half)];
        }
#pragma unroll
        for (int j = 0; j < 4; ++j) {
            const int col = wn * 64 + j * 16 + l15;
#pragma unroll
            for (int v = 0; v < 8; ++v)
                bfm[j].u[v] = *(const unsigned int*)&sB[buf][col][kmap(v, half)];
        }
#pragma unroll
        for (int i = 0; i < 2; ++i)
#pragma unroll
            for (int j = 0; j < 4; ++j)
                acc[i][j] = wmma_bf16(af[i], bfm[j], acc[i][j]);

        if (wid == 0) __builtin_amdgcn_s_wait_tensorcnt(0);
        __syncthreads();
    }

#pragma unroll
    for (int i = 0; i < 2; ++i)
#pragma unroll
        for (int j = 0; j < 4; ++j)
#pragma unroll
            for (int r = 0; r < 8; ++r) {
                const int row = bM + wm * 32 + i * 16 + r + 8 * half;
                const int col = bN + wn * 64 + j * 16 + l15;
                float v = acc[i][j][r];
                if (EPI == 1) v += resid[(size_t)row * N + col];
                C[(size_t)row * N + col] = v;
            }
}

// ---------------- fused W1/W3 GEMM with SwiGLU epilogue -> bf16 ----------------
__global__ __launch_bounds__(256)
void k_gemm_w13_swiglu(const unsigned short* __restrict__ A,
                       const unsigned short* __restrict__ B1,
                       const unsigned short* __restrict__ B3,
                       unsigned short* __restrict__ G,
                       int M, int N, int K) {
    __shared__ __align__(16) unsigned short sA[2][128][40];
    __shared__ __align__(16) unsigned short sB1[2][128][40];
    __shared__ __align__(16) unsigned short sB3[2][128][40];
    const int tid  = threadIdx.x;
    const int lane = tid & 31, wid = tid >> 5;
    const int half = lane >> 4, l15 = lane & 15;
    const int wm = wid & 3, wn = wid >> 2;
    const int bM = blockIdx.y * 128, bN = blockIdx.x * 128;

    const unsigned lA0 = (unsigned)(size_t)&sA[0][0][0];
    const unsigned lA1 = (unsigned)(size_t)&sA[1][0][0];
    const unsigned l10 = (unsigned)(size_t)&sB1[0][0][0];
    const unsigned l11 = (unsigned)(size_t)&sB1[1][0][0];
    const unsigned l30 = (unsigned)(size_t)&sB3[0][0][0];
    const unsigned l31 = (unsigned)(size_t)&sB3[1][0][0];

    v8f au[2][4], av[2][4];
#pragma unroll
    for (int i = 0; i < 2; ++i)
#pragma unroll
        for (int j = 0; j < 4; ++j) { au[i][j] = (v8f)(0.0f); av[i][j] = (v8f)(0.0f); }

    const int KT = K >> 5;
    if (wid == 0) {
        tdm_load_2d(lA0, A  + (size_t)bM * K, 32, 128, (unsigned long long)K, 3, 3, (unsigned)K);
        tdm_load_2d(l10, B1 + (size_t)bN * K, 32, 128, (unsigned long long)K, 3, 3, (unsigned)K);
        tdm_load_2d(l30, B3 + (size_t)bN * K, 32, 128, (unsigned long long)K, 3, 3, (unsigned)K);
        __builtin_amdgcn_s_wait_tensorcnt(0);
    }
    __syncthreads();

    for (int kt = 0; kt < KT; ++kt) {
        const int buf = kt & 1;
        if (wid == 0 && kt + 1 < KT) {
            const int k0 = (kt + 1) << 5;
            tdm_load_2d(buf ? lA0 : lA1, A  + (size_t)bM * K + k0, 32, 128,
                        (unsigned long long)K, 3, 3, (unsigned)K);
            tdm_load_2d(buf ? l10 : l11, B1 + (size_t)bN * K + k0, 32, 128,
                        (unsigned long long)K, 3, 3, (unsigned)K);
            tdm_load_2d(buf ? l30 : l31, B3 + (size_t)bN * K + k0, 32, 128,
                        (unsigned long long)K, 3, 3, (unsigned)K);
        }
        FragBF af[2];
#pragma unroll
        for (int i = 0; i < 2; ++i) {
            const int row = wm * 32 + i * 16 + l15;
#pragma unroll
            for (int v = 0; v < 8; ++v)
                af[i].u[v] = *(const unsigned int*)&sA[buf][row][kmap(v, half)];
        }
        // per-subtile B fragments: keeps VGPR pressure low (no spills)
#pragma unroll
        for (int j = 0; j < 4; ++j) {
            const int col = wn * 64 + j * 16 + l15;
            FragBF b1f, b3f;
#pragma unroll
            for (int v = 0; v < 8; ++v) {
                b1f.u[v] = *(const unsigned int*)&sB1[buf][col][kmap(v, half)];
                b3f.u[v] = *(const unsigned int*)&sB3[buf][col][kmap(v, half)];
            }
#pragma unroll
            for (int i = 0; i < 2; ++i) {
                au[i][j] = wmma_bf16(af[i], b1f, au[i][j]);
                av[i][j] = wmma_bf16(af[i], b3f, av[i][j]);
            }
        }
        if (wid == 0) __builtin_amdgcn_s_wait_tensorcnt(0);
        __syncthreads();
    }
#pragma unroll
    for (int i = 0; i < 2; ++i)
#pragma unroll
        for (int j = 0; j < 4; ++j)
#pragma unroll
            for (int r = 0; r < 8; ++r) {
                const int row = bM + wm * 32 + i * 16 + r + 8 * half;
                const int col = bN + wn * 64 + j * 16 + l15;
                const float u = au[i][j][r];
                const float w = av[i][j][r];
                const float si = u / (1.0f + __expf(-u));     // silu
                G[(size_t)row * N + col] = f2bf(si * w);
            }
}

// ---------------- flash attention (causal) ----------------
__global__ __launch_bounds__(256)
void k_flash_attn(const unsigned short* __restrict__ Q,
                  const unsigned short* __restrict__ Kh,
                  const unsigned short* __restrict__ Vt,
                  unsigned short* __restrict__ O) {
    __shared__ __align__(16) unsigned short sK[32][136];   // 32 keys x 128 dk
    __shared__ __align__(16) unsigned short sV[128][40];   // 128 dk x 32 keys
    __shared__ __align__(16) unsigned short sP[8][16][40]; // per-wave 16x32 P
    const int bh = blockIdx.x;
    const int qb = blockIdx.y;
    const int b  = bh >> 4, h = bh & 15;
    const int tid  = threadIdx.x;
    const int lane = tid & 31, wid = tid >> 5;
    const int half = lane >> 4, l15 = lane & 15;
    const size_t qkBase = (size_t)bh * SEQ * DK;
    const size_t vBase  = (size_t)bh * DK * SEQ;
    const float scale = 0.08838834764831845f;   // 1/sqrt(128)

    const unsigned ldsK = (unsigned)(size_t)&sK[0][0];
    const unsigned ldsV = (unsigned)(size_t)&sV[0][0];

    const int qrow = qb * 128 + wid * 16 + l15;
    FragBF qf[4];
#pragma unroll
    for (int kc = 0; kc < 4; ++kc)
#pragma unroll
        for (int v = 0; v < 8; ++v)
            qf[kc].u[v] = *(const unsigned int*)(Q + qkBase + (size_t)qrow * DK + kc * 32 + kmap(v, half));

    v8f o_acc[8];
    float m_i[8], l_i[8];
#pragma unroll
    for (int j = 0; j < 8; ++j) o_acc[j] = (v8f)(0.0f);
#pragma unroll
    for (int r = 0; r < 8; ++r) { m_i[r] = -3.0e38f; l_i[r] = 0.0f; }

    const int kbEnd = qb * 4 + 4;               // causal truncation
    for (int kb = 0; kb < kbEnd; ++kb) {
        __syncthreads();                        // previous tile fully consumed
        if (wid == 0) {
            // K tile: 32 rows x 128 bf16, row pad 64->68 dwords (interval 64 DW)
            tdm_load_2d(ldsK, Kh + qkBase + (size_t)(kb * 32) * DK,
                        128, 32, (unsigned long long)DK, 5, 3, (unsigned)DK);
            // V tile: 128 rows x 32 bf16 (transposed V), row pad 16->20 dwords
            tdm_load_2d(ldsV, Vt + vBase + kb * 32,
                        32, 128, (unsigned long long)SEQ, 3, 3, (unsigned)SEQ);
            __builtin_amdgcn_s_wait_tensorcnt(0);
        }
        __syncthreads();

        // S = Q * K^T  (two 16x16 tiles over 32 keys)
        v8f s0 = (v8f)(0.0f), s1 = (v8f)(0.0f);
#pragma unroll
        for (int kc = 0; kc < 4; ++kc) {
            FragBF b0, b1;
#pragma unroll
            for (int v = 0; v < 8; ++v) {
                const int kk = kc * 32 + kmap(v, half);
                b0.u[v] = *(const unsigned int*)&sK[l15][kk];
                b1.u[v] = *(const unsigned int*)&sK[16 + l15][kk];
            }
            s0 = wmma_bf16(qf[kc], b0, s0);
            s1 = wmma_bf16(qf[kc], b1, s1);
        }

        // online softmax
#pragma unroll
        for (int r = 0; r < 8; ++r) {
            const int rrow = r + 8 * half;
            const int qi = qb * 128 + wid * 16 + rrow;
            float e0 = s0[r] * scale, e1 = s1[r] * scale;
            const int k0 = kb * 32 + l15, k1 = kb * 32 + 16 + l15;
            if (k0 > qi) e0 = -3.0e38f;
            if (k1 > qi) e1 = -3.0e38f;
            float mx = fmaxf(e0, e1);
            mx = fmaxf(mx, __shfl_xor(mx, 1, 32));
            mx = fmaxf(mx, __shfl_xor(mx, 2, 32));
            mx = fmaxf(mx, __shfl_xor(mx, 4, 32));
            mx = fmaxf(mx, __shfl_xor(mx, 8, 32));
            const float mnew = fmaxf(m_i[r], mx);
            const float p0 = __expf(e0 - mnew);
            const float p1 = __expf(e1 - mnew);
            float rs = p0 + p1;
            rs += __shfl_xor(rs, 1, 32);
            rs += __shfl_xor(rs, 2, 32);
            rs += __shfl_xor(rs, 4, 32);
            rs += __shfl_xor(rs, 8, 32);
            const float fac = __expf(m_i[r] - mnew);
            l_i[r] = l_i[r] * fac + rs;
            m_i[r] = mnew;
#pragma unroll
            for (int j = 0; j < 8; ++j) o_acc[j][r] *= fac;
            sP[wid][rrow][l15]      = f2bf(p0);
            sP[wid][rrow][16 + l15] = f2bf(p1);
        }
        __syncthreads();

        // O += P * V
        FragBF pf;
#pragma unroll
        for (int v = 0; v < 8; ++v)
            pf.u[v] = *(const unsigned int*)&sP[wid][l15][kmap(v, half)];
#pragma unroll
        for (int j = 0; j < 8; ++j) {
            FragBF vf;
#pragma unroll
            for (int v = 0; v < 8; ++v)
                vf.u[v] = *(const unsigned int*)&sV[j * 16 + l15][kmap(v, half)];
            o_acc[j] = wmma_bf16(pf, vf, o_acc[j]);
        }
    }

    // write O (bf16, [B,S,H*dk] layout)
#pragma unroll
    for (int r = 0; r < 8; ++r) {
        const int rrow = r + 8 * half;
        const int srow = qb * 128 + wid * 16 + rrow;
        const float inv = 1.0f / l_i[r];
        const size_t orow = (size_t)(b * SEQ + srow) * D_MODEL + h * DK;
#pragma unroll
        for (int j = 0; j < 8; ++j)
            O[orow + j * 16 + l15] = f2bf(o_acc[j][r] * inv);
    }
}

// ---------------- workspace layout (bytes, all 256-aligned) ----------------
#define OFF_WQKV  ((size_t)0)
#define OFF_WO    ((size_t)25165824)
#define OFF_W1    ((size_t)33554432)
#define OFF_W3    ((size_t)67108864)
#define OFF_W2    ((size_t)100663296)
#define OFF_HBF   ((size_t)134217728)
#define OFF_QKVF  ((size_t)150994944)
#define OFF_QH    ((size_t)251658240)
#define OFF_KH    ((size_t)268435456)
#define OFF_VT    ((size_t)285212672)
#define OFF_OBF   ((size_t)301989888)
#define OFF_X1    ((size_t)318767104)
#define OFF_GBF   ((size_t)352321536)

extern "C" void kernel_launch(void* const* d_in, const int* in_sizes, int n_in,
                              void* d_out, int out_size, void* d_ws, size_t ws_size,
                              hipStream_t stream) {
    const float* x    = (const float*)d_in[0];
    const int*   tpos = (const int*)d_in[1];
    const float* Wqkv = (const float*)d_in[2];
    const float* Wo   = (const float*)d_in[3];
    const float* g1   = (const float*)d_in[4];
    const float* g2   = (const float*)d_in[5];
    const float* W1   = (const float*)d_in[6];
    const float* W3   = (const float*)d_in[7];
    const float* W2   = (const float*)d_in[8];
    float* out = (float*)d_out;
    char*  ws  = (char*)d_ws;

    unsigned short* wqkv_bf = (unsigned short*)(ws + OFF_WQKV);
    unsigned short* wo_bf   = (unsigned short*)(ws + OFF_WO);
    unsigned short* w1_bf   = (unsigned short*)(ws + OFF_W1);
    unsigned short* w3_bf   = (unsigned short*)(ws + OFF_W3);
    unsigned short* w2_bf   = (unsigned short*)(ws + OFF_W2);
    unsigned short* hbf     = (unsigned short*)(ws + OFF_HBF);
    float*          qkvf    = (float*)        (ws + OFF_QKVF);
    unsigned short* qh      = (unsigned short*)(ws + OFF_QH);
    unsigned short* kh      = (unsigned short*)(ws + OFF_KH);
    unsigned short* vt      = (unsigned short*)(ws + OFF_VT);
    unsigned short* obf     = (unsigned short*)(ws + OFF_OBF);
    float*          x1      = (float*)        (ws + OFF_X1);
    unsigned short* gbf     = (unsigned short*)(ws + OFF_GBF);

    k_f32_to_bf16<<<4096, 256, 0, stream>>>(Wqkv, wqkv_bf, D3 * D_MODEL);
    k_f32_to_bf16<<<4096, 256, 0, stream>>>(Wo,   wo_bf,   D_MODEL * D_MODEL);
    k_f32_to_bf16<<<4096, 256, 0, stream>>>(W1,   w1_bf,   D_FF * D_MODEL);
    k_f32_to_bf16<<<4096, 256, 0, stream>>>(W3,   w3_bf,   D_FF * D_MODEL);
    k_f32_to_bf16<<<4096, 256, 0, stream>>>(W2,   w2_bf,   D_MODEL * D_FF);

    k_rmsnorm_bf16<<<NTOK, 256, 0, stream>>>(x, g1, hbf);

    k_gemm_bf16<0><<<dim3(D3 / 128, NTOK / 128), 256, 0, stream>>>(
        hbf, wqkv_bf, qkvf, nullptr, NTOK, D3, D_MODEL);

    k_rope_pack<<<(BATCH * N_HEADS * SEQ * (DK / 2)) / 256, 256, 0, stream>>>(
        qkvf, tpos, qh, kh, vt);

    k_flash_attn<<<dim3(BATCH * N_HEADS, SEQ / 128), 256, 0, stream>>>(qh, kh, vt, obf);

    k_gemm_bf16<1><<<dim3(D_MODEL / 128, NTOK / 128), 256, 0, stream>>>(
        obf, wo_bf, x1, x, NTOK, D_MODEL, D_MODEL);

    k_rmsnorm_bf16<<<NTOK, 256, 0, stream>>>(x1, g2, hbf);

    k_gemm_w13_swiglu<<<dim3(D_FF / 128, NTOK / 128), 256, 0, stream>>>(
        hbf, w1_bf, w3_bf, gbf, NTOK, D_FF, D_MODEL);

    k_gemm_bf16<1><<<dim3(D_MODEL / 128, NTOK / 128), 256, 0, stream>>>(
        gbf, w2_bf, out, x1, NTOK, D_MODEL, D_FF);
}
